// TARGCN_cell_40982577938733
// MI455X (gfx1250) — matrix operand order repaired and checked by reading, not compiled, not run on hardware
//
#include <hip/hip_runtime.h>

// ---------------------------------------------------------------------------
// dims
// ---------------------------------------------------------------------------
constexpr int  Bsz  = 128;
constexpr int  Tt   = 12;
constexpr int  Nn   = 307;
constexpr int  Dd   = 64;
constexpr int  EMB  = 10;
constexpr int  CV   = 65;            // DIM_IN + D (valid channels)
constexpr int  NP   = 320;           // nodes padded (mult of 32)
constexpr int  CP   = 96;            // channels padded (mult of 32)
constexpr int  KC   = 2 * CP;        // 192: cheb-K * padded channels
constexpr int  NCOL = Bsz * CP;      // 12288 columns of the graph-mix GEMM
constexpr long Rseq = (long)Nn * Bsz;   // 39296 sequences
constexpr long RT   = Rseq * Tt;        // 471552 tokens

typedef __attribute__((ext_vector_type(16))) __bf16 v16bf;
typedef __attribute__((ext_vector_type(2)))  __bf16 v2bf;
typedef __attribute__((ext_vector_type(8)))  float  v8f;

// ---------------------------------------------------------------------------
// WMMA fragment helpers (layouts per CDNA5 ISA 7.12.2, wave32)
// ---------------------------------------------------------------------------
__device__ __forceinline__ v8f vzero8() {
  v8f z;
#pragma unroll
  for (int i = 0; i < 8; ++i) z[i] = 0.f;
  return z;
}

// A operand 16x32 bf16: lane<16 -> M=lane, K in {h*8..h*8+7} U {16+h*8..}
template <typename P>
__device__ __forceinline__ v16bf frag_a(P p, int ld) {
  const int lane = threadIdx.x & 31;
  const int half = lane >> 4;
  const int m    = lane & 15;
  v16bf f;
#pragma unroll
  for (int j = 0; j < 8; ++j) f[j] = p[m * ld + half * 8 + j];
#pragma unroll
  for (int j = 0; j < 8; ++j) f[8 + j] = p[m * ld + 16 + half * 8 + j];
  return f;
}

// B operand 32x16 bf16 from a K-contiguous ("transposed") buffer:
// lane<16 -> N=lane, K in {0..7} U {16..23} ; lane>=16 -> N=lane-16, K in {8..15} U {24..31}
template <typename P>
__device__ __forceinline__ v16bf frag_b(P p, int ld) {
  const int lane = threadIdx.x & 31;
  const int off  = (lane & 15) * ld + (lane >> 4) * 8;
  v16bf f;
#pragma unroll
  for (int j = 0; j < 8; ++j) f[j] = p[off + j];
#pragma unroll
  for (int j = 0; j < 8; ++j) f[8 + j] = p[off + 16 + j];
  return f;
}

__device__ __forceinline__ v8f wmma_bf16(v16bf a, v16bf b, v8f c) {
  return __builtin_amdgcn_wmma_f32_16x16x32_bf16(false, a, false, b, (short)0, c,
                                                 false, false);
}

__device__ __forceinline__ float sigmoidf_(float x) {
  return 1.f / (1.f + __expf(-x));
}

// ---------------------------------------------------------------------------
// gfx1250 async global->LDS copy (ASYNCcnt path), 16B per lane per issue
// ---------------------------------------------------------------------------
__device__ __forceinline__ void async_copy16(unsigned lds_off, const void* gptr) {
  asm volatile("global_load_async_to_lds_b128 %0, %1, off"
               :
               : "v"(lds_off), "v"(gptr)
               : "memory");
}
__device__ __forceinline__ void wait_async0() {
  asm volatile("s_wait_asynccnt 0x0" ::: "memory");
}
__device__ __forceinline__ unsigned lds_off_of(const void* p) {
  return (unsigned)(unsigned long long)p;  // low 32 bits of flat LDS address
}

// ---------------------------------------------------------------------------
// adjacency: A = softmax(relu(E E^T)) rows, written bf16 into padded [NP][NP]
// ---------------------------------------------------------------------------
__global__ void k_adj(const float* __restrict__ E, __bf16* __restrict__ Abf) {
  __shared__ float s[Nn];
  __shared__ float red[128];
  const int n   = blockIdx.x;
  const int tid = threadIdx.x;
  float e[EMB];
#pragma unroll
  for (int d = 0; d < EMB; ++d) e[d] = E[n * EMB + d];
  for (int m = tid; m < Nn; m += 128) {
    float acc = 0.f;
#pragma unroll
    for (int d = 0; d < EMB; ++d) acc += e[d] * E[m * EMB + d];
    s[m] = fmaxf(acc, 0.f);
  }
  __syncthreads();
  float mx = -1e30f;
  for (int m = tid; m < Nn; m += 128) mx = fmaxf(mx, s[m]);
  red[tid] = mx;
  __syncthreads();
  for (int st = 64; st > 0; st >>= 1) {
    if (tid < st) red[tid] = fmaxf(red[tid], red[tid + st]);
    __syncthreads();
  }
  mx = red[0];
  __syncthreads();
  float sum = 0.f;
  for (int m = tid; m < Nn; m += 128) {
    float v = __expf(s[m] - mx);
    s[m]    = v;
    sum += v;
  }
  red[tid] = sum;
  __syncthreads();
  for (int st = 64; st > 0; st >>= 1) {
    if (tid < st) red[tid] += red[tid + st];
    __syncthreads();
  }
  const float inv = 1.f / red[0];
  for (int m = tid; m < Nn; m += 128)
    Abf[(long)n * NP + m] = (__bf16)(s[m] * inv);
}

// ---------------------------------------------------------------------------
// node-adaptive weights: WT[n][o][k*CP+c] = sum_d E[n,d] pool[d][k][c][o] (bf16)
// ---------------------------------------------------------------------------
__global__ void k_make_w(const float* __restrict__ E, const float* __restrict__ pool,
                         __bf16* __restrict__ WT, int Odim) {
  long idx         = (long)blockIdx.x * blockDim.x + threadIdx.x;
  const long total = (long)Nn * Odim * 2 * CV;
  if (idx >= total) return;
  int  c = (int)(idx % CV);
  long t = idx / CV;
  int  k = (int)(t & 1);
  t >>= 1;
  int n = (int)(t / Odim);
  int o = (int)(t % Odim);
  float acc = 0.f;
#pragma unroll
  for (int d = 0; d < EMB; ++d)
    acc += E[n * EMB + d] * pool[(((long)d * 2 + k) * CV + c) * Odim + o];
  WT[((long)n * Odim + o) * KC + k * CP + c] = (__bf16)acc;
}

__global__ void k_make_b(const float* __restrict__ E, const float* __restrict__ bpool,
                         float* __restrict__ Bo, int Odim) {
  long idx = (long)blockIdx.x * blockDim.x + threadIdx.x;
  if (idx >= (long)Nn * Odim) return;
  int   n   = (int)(idx / Odim);
  int   o   = (int)(idx % Odim);
  float acc = 0.f;
#pragma unroll
  for (int d = 0; d < EMB; ++d) acc += E[n * EMB + d] * bpool[d * Odim + o];
  Bo[idx] = acc;
}

// ---------------------------------------------------------------------------
// h init: H[n][b][d] = init_state[0][b][n][d]
// ---------------------------------------------------------------------------
__global__ void k_init_h(const float* __restrict__ h0, float* __restrict__ H) {
  long idx = (long)blockIdx.x * blockDim.x + threadIdx.x;
  if (idx >= (long)Nn * Bsz * Dd) return;
  int d = (int)(idx & 63);
  long r = idx >> 6;
  int b = (int)(r % Bsz);
  int n = (int)(r / Bsz);
  H[idx] = h0[((long)b * Nn + n) * Dd + d];
}

// ---------------------------------------------------------------------------
// assemble xs = [x_t, h]  ->  XSn [n][b][c]  and  XST [b*CP+c][node]
// ---------------------------------------------------------------------------
__global__ void k_assemble_xs(const float* __restrict__ x, const float* __restrict__ H,
                              __bf16* __restrict__ XSn, __bf16* __restrict__ XST,
                              int t) {
  long idx = (long)blockIdx.x * blockDim.x + threadIdx.x;
  if (idx >= Rseq) return;
  const int b = (int)(idx % Bsz);
  const int n = (int)(idx / Bsz);
  __bf16* pn  = XSn + ((long)n * Bsz + b) * CP;
  __bf16* pt  = XST + (long)b * CP * NP + n;
  const float xv = x[((long)b * Tt + t) * Nn + n];
  pn[0] = (__bf16)xv;
  pt[0] = (__bf16)xv;
  const float* h = H + ((long)n * Bsz + b) * Dd;
#pragma unroll 4
  for (int d = 0; d < Dd; ++d) {
    __bf16 v        = (__bf16)h[d];
    pn[1 + d]       = v;
    pt[(long)(1 + d) * NP] = v;
  }
}

// assemble cand = [x_t, z*h]
__global__ void k_assemble_cand(const float* __restrict__ x, const float* __restrict__ H,
                                const float* __restrict__ Z, __bf16* __restrict__ XSn,
                                __bf16* __restrict__ XST, int t) {
  long idx = (long)blockIdx.x * blockDim.x + threadIdx.x;
  if (idx >= Rseq) return;
  const int b = (int)(idx % Bsz);
  const int n = (int)(idx / Bsz);
  __bf16* pn  = XSn + ((long)n * Bsz + b) * CP;
  __bf16* pt  = XST + (long)b * CP * NP + n;
  const float xv = x[((long)b * Tt + t) * Nn + n];
  pn[0] = (__bf16)xv;
  pt[0] = (__bf16)xv;
  const float* h = H + ((long)n * Bsz + b) * Dd;
  const float* z = Z + ((long)n * Bsz + b) * Dd;
#pragma unroll 4
  for (int d = 0; d < Dd; ++d) {
    __bf16 v        = (__bf16)(z[d] * h[d]);
    pn[1 + d]       = v;
    pt[(long)(1 + d) * NP] = v;
  }
}

// ---------------------------------------------------------------------------
// graph mix: XM[node][col] = sum_m A[node][m] * XT[col][m]
// one wave per 16x64 D tile, software-pipelined one K-chunk ahead so the
// matrix pipe never drains loadcnt to 0; output staged through LDS and
// written as coalesced b128 rows. A/XT are L2-resident.
// ---------------------------------------------------------------------------
__global__ void k_graph_mix(const __bf16* __restrict__ Abf,
                            const __bf16* __restrict__ XT,
                            __bf16* __restrict__ XM) {
  __shared__ __align__(16) __bf16 sOut[4][16 * 64];  // 8 KB
  const int wave = blockIdx.x * 4 + (threadIdx.x >> 5);
  const int mt   = wave / (NCOL / 64);
  const int cg   = wave % (NCOL / 64);  // 64-column group
  const int lane = threadIdx.x & 31;
  v8f acc[4];
#pragma unroll
  for (int i = 0; i < 4; ++i) acc[i] = vzero8();
  const __bf16* ap = Abf + (long)mt * 16 * NP;
  const __bf16* bp = XT + (long)cg * 64 * NP;

  // prologue: load chunk 0
  v16bf a  = frag_a(ap, NP);
  v16bf b0 = frag_b(bp + 0 * 16 * NP, NP);
  v16bf b1 = frag_b(bp + 1 * 16 * NP, NP);
  v16bf b2 = frag_b(bp + 2 * 16 * NP, NP);
  v16bf b3 = frag_b(bp + 3 * 16 * NP, NP);
#pragma unroll
  for (int kk = 32; kk < NP + 32; kk += 32) {
    v16bf an, bn0, bn1, bn2, bn3;
    if (kk < NP) {  // prefetch next chunk before consuming current one
      an  = frag_a(ap + kk, NP);
      bn0 = frag_b(bp + 0 * 16 * NP + kk, NP);
      bn1 = frag_b(bp + 1 * 16 * NP + kk, NP);
      bn2 = frag_b(bp + 2 * 16 * NP + kk, NP);
      bn3 = frag_b(bp + 3 * 16 * NP + kk, NP);
    }
    acc[0] = wmma_bf16(a, b0, acc[0]);
    acc[1] = wmma_bf16(a, b1, acc[1]);
    acc[2] = wmma_bf16(a, b2, acc[2]);
    acc[3] = wmma_bf16(a, b3, acc[3]);
    if (kk < NP) {
      a  = an;
      b0 = bn0;
      b1 = bn1;
      b2 = bn2;
      b3 = bn3;
    }
  }
  const int half = lane >> 4;
  __bf16*   st   = sOut[threadIdx.x >> 5];
#pragma unroll
  for (int c = 0; c < 4; ++c)
#pragma unroll
    for (int r = 0; r < 8; ++r)
      st[(r + 8 * half) * 64 + c * 16 + (lane & 15)] = (__bf16)acc[c][r];
  __syncthreads();
  // 16 rows x 128B per wave tile -> 4 int4 stores per lane
  const long base = (long)(mt * 16) * NCOL + cg * 64;
#pragma unroll
  for (int i = 0; i < 4; ++i) {
    const int idx = i * 32 + lane;
    const int row = idx >> 3;
    const int ck  = idx & 7;
    *(int4*)(XM + base + (long)row * NCOL + ck * 8) = ((const int4*)st)[idx];
  }
}

// ---------------------------------------------------------------------------
// per-node gate GEMM: [B=128,K=192] @ [192,128] + bias -> sigmoid -> Z,R
// weight slab streamed into LDS via async global->LDS (ASYNCcnt) loads
// ---------------------------------------------------------------------------
__global__ void k_node_gate(const __bf16* __restrict__ XSn,
                            const __bf16* __restrict__ XM,
                            const __bf16* __restrict__ WgT,
                            const float* __restrict__ Bg, float* __restrict__ Z,
                            float* __restrict__ Rb) {
  __shared__ __align__(16) __bf16 sW[128 * KC];  // 48 KB node weight slab
  const int n = blockIdx.x;
  {
    const unsigned sb   = lds_off_of(&sW[0]);
    const char*    gsrc = (const char*)(WgT + (long)n * 128 * KC);
    for (int i = threadIdx.x; i < 128 * KC * 2 / 16; i += 256)
      async_copy16(sb + i * 16, gsrc + (long)i * 16);
    wait_async0();
  }
  __syncthreads();
  const int wv   = threadIdx.x >> 5;
  const int lane = threadIdx.x & 31;
  v8f acc[8];
#pragma unroll
  for (int i = 0; i < 8; ++i) acc[i] = vzero8();
  const __bf16* Xa = XSn + ((long)n * Bsz + wv * 16) * CP;
  const __bf16* Xm = XM + (long)n * NCOL + (long)(wv * 16) * CP;
#pragma unroll
  for (int kit = 0; kit < 6; ++kit) {
    const __bf16* ap = (kit < 3) ? (Xa + kit * 32) : (Xm + (kit - 3) * 32);
    v16bf a = frag_a(ap, CP);
#pragma unroll
    for (int nt = 0; nt < 8; ++nt) {
      v16bf b = frag_b(&sW[0] + nt * 16 * KC + kit * 32, KC);
      acc[nt] = wmma_bf16(a, b, acc[nt]);
    }
  }
  const int half = lane >> 4;
#pragma unroll
  for (int nt = 0; nt < 8; ++nt) {
    const int o = nt * 16 + (lane & 15);
#pragma unroll
    for (int r = 0; r < 8; ++r) {
      const int   b = wv * 16 + r + 8 * half;
      const float v = sigmoidf_(acc[nt][r] + Bg[n * 128 + o]);
      if (o < Dd)
        Z[((long)n * Bsz + b) * Dd + o] = v;
      else
        Rb[((long)n * Bsz + b) * Dd + (o - Dd)] = v;
    }
  }
}

// ---------------------------------------------------------------------------
// per-node update GEMM + tanh + GRU combine; writes H and state history Y
// ---------------------------------------------------------------------------
__global__ void k_node_update(const __bf16* __restrict__ XSn,
                              const __bf16* __restrict__ XM,
                              const __bf16* __restrict__ WuT,
                              const float* __restrict__ Bu,
                              const float* __restrict__ Rb, float* __restrict__ H,
                              float* __restrict__ Y, int t) {
  __shared__ __align__(16) __bf16 sW[64 * KC];  // 24 KB
  const int n = blockIdx.x;
  {
    const unsigned sb   = lds_off_of(&sW[0]);
    const char*    gsrc = (const char*)(WuT + (long)n * 64 * KC);
    for (int i = threadIdx.x; i < 64 * KC * 2 / 16; i += 256)
      async_copy16(sb + i * 16, gsrc + (long)i * 16);
    wait_async0();
  }
  __syncthreads();
  const int wv   = threadIdx.x >> 5;
  const int lane = threadIdx.x & 31;
  v8f acc[4];
#pragma unroll
  for (int i = 0; i < 4; ++i) acc[i] = vzero8();
  const __bf16* Xa = XSn + ((long)n * Bsz + wv * 16) * CP;
  const __bf16* Xm = XM + (long)n * NCOL + (long)(wv * 16) * CP;
#pragma unroll
  for (int kit = 0; kit < 6; ++kit) {
    const __bf16* ap = (kit < 3) ? (Xa + kit * 32) : (Xm + (kit - 3) * 32);
    v16bf a = frag_a(ap, CP);
#pragma unroll
    for (int nt = 0; nt < 4; ++nt) {
      v16bf b = frag_b(&sW[0] + nt * 16 * KC + kit * 32, KC);
      acc[nt] = wmma_bf16(a, b, acc[nt]);
    }
  }
  const int half = lane >> 4;
#pragma unroll
  for (int nt = 0; nt < 4; ++nt) {
    const int o = nt * 16 + (lane & 15);
#pragma unroll
    for (int r = 0; r < 8; ++r) {
      const int   b    = wv * 16 + r + 8 * half;
      const long  base = ((long)n * Bsz + b) * Dd + o;
      const float hc   = tanhf(acc[nt][r] + Bu[n * Dd + o]);
      const float rr   = Rb[base];
      const float hn   = rr * H[base] + (1.f - rr) * hc;
      H[base]          = hn;
      Y[(((long)n * Bsz + b) * Tt + t) * Dd + o] = hn;
    }
  }
}

// ---------------------------------------------------------------------------
// f32 -> bf16 bulk convert
// ---------------------------------------------------------------------------
__global__ void k_f2bf(const float* __restrict__ src, __bf16* __restrict__ dst,
                       long total) {
  long idx = (long)blockIdx.x * blockDim.x + threadIdx.x;
  if (idx < total) dst[idx] = (__bf16)src[idx];
}

// ---------------------------------------------------------------------------
// QKV projection: [RT,64] @ [64,192] -> QKV bf16 [RT][192]
// both global A fragments hoisted ahead of the WMMA chain; output tile staged
// in LDS (reusing the weight slab) -> contiguous b128 burst
// ---------------------------------------------------------------------------
__global__ void k_qkv(const __bf16* __restrict__ Ybf, const float* __restrict__ Wq,
                      const float* __restrict__ Wk, const float* __restrict__ Wv,
                      __bf16* __restrict__ QKV) {
  __shared__ __align__(16) __bf16 smem[128 * 192];  // 48 KB (W: first 12288)
  __bf16* sW = smem;
  for (int i = threadIdx.x; i < 192 * 64; i += 256) {
    const int o = i >> 6, k = i & 63;
    const float* W = (o < 64) ? Wq : ((o < 128) ? Wk : Wv);
    sW[i]          = (__bf16)W[k * 64 + (o & 63)];
  }
  __syncthreads();
  const int  wv   = threadIdx.x >> 5;
  const int  lane = threadIdx.x & 31;
  const long mrow = ((long)blockIdx.x * 8 + wv) * 16;
  v8f acc[12];
#pragma unroll
  for (int i = 0; i < 12; ++i) acc[i] = vzero8();
  const __bf16* Ap = Ybf + mrow * 64;
  v16bf a0 = frag_a(Ap, 64);        // issue both long-latency loads up front
  v16bf a1 = frag_a(Ap + 32, 64);
#pragma unroll
  for (int nt = 0; nt < 12; ++nt) {
    v16bf b = frag_b(&sW[0] + nt * 16 * 64, 64);
    acc[nt] = wmma_bf16(a0, b, acc[nt]);
  }
#pragma unroll
  for (int nt = 0; nt < 12; ++nt) {
    v16bf b = frag_b(&sW[0] + nt * 16 * 64 + 32, 64);
    acc[nt] = wmma_bf16(a1, b, acc[nt]);
  }
  __syncthreads();  // done reading sW; reuse smem as output staging
  const int half = lane >> 4;
#pragma unroll
  for (int nt = 0; nt < 12; ++nt) {
    const int col = nt * 16 + (lane & 15);
#pragma unroll
    for (int r = 0; r < 8; ++r)
      smem[(wv * 16 + r + 8 * half) * 192 + col] = (__bf16)acc[nt][r];
  }
  __syncthreads();
  const long gbase = (long)blockIdx.x * 128 * 192;
  int4*      gdst  = (int4*)(QKV + gbase);
  for (int i = threadIdx.x; i < 128 * 192 * 2 / 16; i += 256)
    gdst[i] = ((const int4*)smem)[i];
}

// ---------------------------------------------------------------------------
// attention core: one wave per sequence (T=12, 2 heads of 32) — tiny, VALU+LDS;
// Q/K/V rows pulled in via async global->LDS loads
// ---------------------------------------------------------------------------
__global__ void k_attn(const __bf16* __restrict__ QKV, __bf16* __restrict__ Obf) {
  __shared__ float  sS[4][Tt][Tt + 1];
  __shared__ __align__(16) __bf16 sQ[4][Tt][192];
  const int  wv   = threadIdx.x >> 5;
  const int  lane = threadIdx.x & 31;
  const long s    = (long)blockIdx.x * 4 + wv;
  {
    const unsigned sb   = lds_off_of(&sQ[wv][0][0]);
    const char*    gsrc = (const char*)(QKV + s * Tt * 192);
    for (int i = lane; i < Tt * 192 * 2 / 16; i += 32)
      async_copy16(sb + i * 16, gsrc + (long)i * 16);
    wait_async0();
  }
  __syncthreads();
  const float scale = 0.1767766952966369f;  // 1/sqrt(32)
  for (int h = 0; h < 2; ++h) {
    for (int p = lane; p < Tt * Tt; p += 32) {
      const int t0 = p / Tt, u = p % Tt;
      float a = 0.f;
#pragma unroll
      for (int d = 0; d < 32; ++d)
        a += (float)sQ[wv][t0][h * 32 + d] * (float)sQ[wv][u][64 + h * 32 + d];
      sS[wv][t0][u] = a * scale;
    }
    __syncthreads();
    if (lane < Tt) {
      float mx = -1e30f;
#pragma unroll
      for (int u = 0; u < Tt; ++u) mx = fmaxf(mx, sS[wv][lane][u]);
      float sum = 0.f;
#pragma unroll
      for (int u = 0; u < Tt; ++u) {
        float v = __expf(sS[wv][lane][u] - mx);
        sS[wv][lane][u] = v;
        sum += v;
      }
      const float inv = 1.f / sum;
#pragma unroll
      for (int u = 0; u < Tt; ++u) sS[wv][lane][u] *= inv;
    }
    __syncthreads();
    for (int p = lane; p < Tt * 32; p += 32) {
      const int t0 = p >> 5, dc = p & 31;
      float o = 0.f;
#pragma unroll
      for (int u = 0; u < Tt; ++u)
        o += sS[wv][t0][u] * (float)sQ[wv][u][128 + h * 32 + dc];
      Obf[(s * Tt + t0) * Dd + h * 32 + dc] = (__bf16)o;
    }
    __syncthreads();
  }
}

// ---------------------------------------------------------------------------
// out projection + residual + layernorm; updates Y (f32) and Ybf (bf16)
// ---------------------------------------------------------------------------
__global__ void k_out_ln(const __bf16* __restrict__ Obf, const float* __restrict__ Wo,
                         const float* __restrict__ g, const float* __restrict__ be,
                         float* __restrict__ Y, __bf16* __restrict__ Ybf) {
  __shared__ __bf16 sW[64 * 64];       // transposed Wo, 8 KB
  __shared__ float  sA[128][Dd + 1];   // GEMM result tile
  for (int i = threadIdx.x; i < 64 * 64; i += 256) {
    const int o = i >> 6, k = i & 63;
    sW[i] = (__bf16)Wo[k * 64 + o];
  }
  __syncthreads();
  const int  wv   = threadIdx.x >> 5;
  const int  lane = threadIdx.x & 31;
  const long mrow = ((long)blockIdx.x * 8 + wv) * 16;
  v8f acc[4];
#pragma unroll
  for (int i = 0; i < 4; ++i) acc[i] = vzero8();
  const __bf16* Ap = Obf + mrow * 64;
  v16bf a0 = frag_a(Ap, 64);
  v16bf a1 = frag_a(Ap + 32, 64);
#pragma unroll
  for (int nt = 0; nt < 4; ++nt) {
    v16bf b = frag_b(&sW[0] + nt * 16 * 64, 64);
    acc[nt] = wmma_bf16(a0, b, acc[nt]);
  }
#pragma unroll
  for (int nt = 0; nt < 4; ++nt) {
    v16bf b = frag_b(&sW[0] + nt * 16 * 64 + 32, 64);
    acc[nt] = wmma_bf16(a1, b, acc[nt]);
  }
  const int half = lane >> 4;
#pragma unroll
  for (int nt = 0; nt < 4; ++nt)
#pragma unroll
    for (int r = 0; r < 8; ++r)
      sA[wv * 16 + r + 8 * half][nt * 16 + (lane & 15)] = acc[nt][r];
  __syncthreads();
  if (threadIdx.x < 128) {
    const long row  = (long)blockIdx.x * 128 + threadIdx.x;
    float*     yrow = Y + row * Dd;
    float tmp[Dd];
    float mean = 0.f;
#pragma unroll
    for (int d = 0; d < Dd; ++d) {
      const float v = yrow[d] + sA[threadIdx.x][d];
      tmp[d]        = v;
      mean += v;
    }
    mean *= (1.f / Dd);
    float var = 0.f;
#pragma unroll
    for (int d = 0; d < Dd; ++d) {
      const float c = tmp[d] - mean;
      var += c * c;
    }
    var *= (1.f / Dd);
    const float rstd = rsqrtf(var + 1e-5f);
    float4* y4  = (float4*)yrow;
    v2bf*   yb2 = (v2bf*)(Ybf + row * Dd);
#pragma unroll
    for (int d = 0; d < Dd; d += 4) {
      float4 o4;
      o4.x = (tmp[d + 0] - mean) * rstd * g[d + 0] + be[d + 0];
      o4.y = (tmp[d + 1] - mean) * rstd * g[d + 1] + be[d + 1];
      o4.z = (tmp[d + 2] - mean) * rstd * g[d + 2] + be[d + 2];
      o4.w = (tmp[d + 3] - mean) * rstd * g[d + 3] + be[d + 3];
      y4[d >> 2] = o4;
      v2bf p0;
      p0[0] = (__bf16)o4.x;
      p0[1] = (__bf16)o4.y;
      yb2[(d >> 1) + 0] = p0;
      v2bf p1;
      p1[0] = (__bf16)o4.z;
      p1[1] = (__bf16)o4.w;
      yb2[(d >> 1) + 1] = p1;
    }
  }
}

// ---------------------------------------------------------------------------
// final: out[b][t][n][d] = Y[(n*B+b)*T+t][d] ; last_state[b][n][d] = H[n][b][d]
// ---------------------------------------------------------------------------
__global__ void k_final(const float* __restrict__ Y, const float* __restrict__ H,
                        float* __restrict__ out, long total) {
  long idx = (long)blockIdx.x * blockDim.x + threadIdx.x;
  if (idx >= total) return;
  const long OUT1 = (long)Bsz * Tt * Nn * Dd;
  if (idx < OUT1) {
    const int d = (int)(idx & 63);
    long r = idx >> 6;
    const int n = (int)(r % Nn);
    r /= Nn;
    const int t = (int)(r % Tt);
    const int b = (int)(r / Tt);
    out[idx] = Y[(((long)n * Bsz + b) * Tt + t) * Dd + d];
  } else {
    long j = idx - OUT1;
    const int d = (int)(j & 63);
    long r = j >> 6;
    const int n = (int)(r % Nn);
    const int b = (int)(r / Nn);
    out[idx] = H[((long)n * Bsz + b) * Dd + d];
  }
}

// ---------------------------------------------------------------------------
// host
// ---------------------------------------------------------------------------
static inline unsigned cdivu(long a, long b) { return (unsigned)((a + b - 1) / b); }

extern "C" void kernel_launch(void* const* d_in, const int* in_sizes, int n_in,
                              void* d_out, int out_size, void* d_ws, size_t ws_size,
                              hipStream_t stream) {
  (void)in_sizes; (void)n_in; (void)out_size; (void)ws_size;
  const float* x   = (const float*)d_in[0];
  const float* h0  = (const float*)d_in[1];
  const float* E   = (const float*)d_in[2];
  const float* gwp = (const float*)d_in[3];
  const float* gbp = (const float*)d_in[4];
  const float* uwp = (const float*)d_in[5];
  const float* ubp = (const float*)d_in[6];
  const float* wq  = (const float*)d_in[7];
  const float* wk  = (const float*)d_in[8];
  const float* wv  = (const float*)d_in[9];
  const float* wo  = (const float*)d_in[10];
  const float* lng = (const float*)d_in[11];
  const float* lnb = (const float*)d_in[12];
  float* out = (float*)d_out;

  char*  ws  = (char*)d_ws;
  size_t off = 0;
  auto alloc = [&](size_t bytes) -> char* {
    char* p = ws + off;
    off += (bytes + 255) & ~(size_t)255;
    return p;
  };
  __bf16* Abf = (__bf16*)alloc((size_t)NP * NP * 2);
  __bf16* WgT = (__bf16*)alloc((size_t)Nn * 128 * KC * 2);
  __bf16* WuT = (__bf16*)alloc((size_t)Nn * 64 * KC * 2);
  float*  Bg  = (float*)alloc((size_t)Nn * 128 * 4);
  float*  Bu  = (float*)alloc((size_t)Nn * 64 * 4);
  __bf16* XSn = (__bf16*)alloc((size_t)NP * Bsz * CP * 2);
  __bf16* XST = (__bf16*)alloc((size_t)NCOL * NP * 2);
  __bf16* XM  = (__bf16*)alloc((size_t)NP * NCOL * 2);
  float*  Z   = (float*)alloc((size_t)Nn * Bsz * Dd * 4);
  float*  Rb  = (float*)alloc((size_t)Nn * Bsz * Dd * 4);
  float*  H   = (float*)alloc((size_t)Nn * Bsz * Dd * 4);
  float*  Y   = (float*)alloc((size_t)RT * Dd * 4);
  __bf16* Ybf = (__bf16*)alloc((size_t)RT * Dd * 2);
  __bf16* QKV = (__bf16*)alloc((size_t)RT * 192 * 2);
  __bf16* Obf = (__bf16*)alloc((size_t)RT * Dd * 2);

  // zero the padded regions once per launch (graph-capture safe)
  hipMemsetAsync(Abf, 0, (size_t)NP * NP * 2, stream);
  hipMemsetAsync(WgT, 0, (size_t)Nn * 128 * KC * 2, stream);
  hipMemsetAsync(WuT, 0, (size_t)Nn * 64 * KC * 2, stream);
  hipMemsetAsync(XSn, 0, (size_t)NP * Bsz * CP * 2, stream);
  hipMemsetAsync(XST, 0, (size_t)NCOL * NP * 2, stream);

  k_adj<<<Nn, 128, 0, stream>>>(E, Abf);
  k_make_w<<<cdivu((long)Nn * 128 * 2 * CV, 256), 256, 0, stream>>>(E, gwp, WgT, 128);
  k_make_w<<<cdivu((long)Nn * 64 * 2 * CV, 256), 256, 0, stream>>>(E, uwp, WuT, 64);
  k_make_b<<<cdivu((long)Nn * 128, 256), 256, 0, stream>>>(E, gbp, Bg, 128);
  k_make_b<<<cdivu((long)Nn * 64, 256), 256, 0, stream>>>(E, ubp, Bu, 64);
  k_init_h<<<cdivu((long)Nn * Bsz * Dd, 256), 256, 0, stream>>>(h0, H);

  const unsigned gmix_blocks = (NP / 16) * (NCOL / 64) / 4;  // 960
  for (int t = 0; t < Tt; ++t) {
    k_assemble_xs<<<cdivu(Rseq, 128), 128, 0, stream>>>(x, H, XSn, XST, t);
    k_graph_mix<<<gmix_blocks, 128, 0, stream>>>(Abf, XST, XM);
    k_node_gate<<<Nn, 256, 0, stream>>>(XSn, XM, WgT, Bg, Z, Rb);
    k_assemble_cand<<<cdivu(Rseq, 128), 128, 0, stream>>>(x, H, Z, XSn, XST, t);
    k_graph_mix<<<gmix_blocks, 128, 0, stream>>>(Abf, XST, XM);
    k_node_update<<<Nn, 256, 0, stream>>>(XSn, XM, WuT, Bu, Rb, H, Y, t);
  }

  k_f2bf<<<cdivu(RT * Dd, 256), 256, 0, stream>>>(Y, Ybf, RT * Dd);
  for (int l = 0; l < 2; ++l) {
    k_qkv<<<(unsigned)(RT / 128), 256, 0, stream>>>(Ybf, wq + (long)l * 4096,
                                                    wk + (long)l * 4096,
                                                    wv + (long)l * 4096, QKV);
    k_attn<<<(unsigned)(Rseq / 4), 128, 0, stream>>>(QKV, Obf);
    k_out_ln<<<(unsigned)(RT / 128), 256, 0, stream>>>(Obf, wo + (long)l * 4096,
                                                       lng + l * 64, lnb + l * 64, Y,
                                                       Ybf);
  }
  const long total = (long)Bsz * Tt * Nn * Dd + (long)Bsz * Nn * Dd;
  k_final<<<cdivu(total, 256), 256, 0, stream>>>(Y, H, out, total);
}
